// LocalAttnLayer_49143015801165
// MI455X (gfx1250) — matrix operand order repaired and checked by reading, not compiled, not run on hardware
//
#include <hip/hip_runtime.h>
#include <hip/hip_bf16.h>
#include <cstdint>
#include <cstddef>

#define D_MODEL 1024
#define N_HEADS 16
#define HEAD_DIM 64
#define WINDOW 128
#define FF_HID 2048
#define BATCH 4
#define SEQ 4096
#define BS (BATCH * SEQ)      /* 16384 rows */
#define BHT (BATCH * N_HEADS) /* 64 */
#define NW (SEQ / WINDOW)     /* 32 windows */

typedef __attribute__((ext_vector_type(16))) _Float16 v16h;
typedef __attribute__((ext_vector_type(8)))  _Float16 v8h;
typedef __attribute__((ext_vector_type(4)))  _Float16 v4h;
typedef __attribute__((ext_vector_type(8)))  float    v8f;
typedef __attribute__((ext_vector_type(4)))  float    v4f;

union F16Frag { v16h v; v8h h[2]; };

__device__ inline v8f zero8() {
  v8f z;
#pragma unroll
  for (int i = 0; i < 8; ++i) z[i] = 0.0f;
  return z;
}

__device__ inline float waveReduceAdd(float v) {
#pragma unroll
  for (int m = 1; m < 32; m <<= 1) v += __shfl_xor(v, m);
  return v;
}

// ---------------------------------------------------------------------------
// Elementwise converts
// ---------------------------------------------------------------------------
__global__ void f32_to_f16_kernel(const float* __restrict__ in,
                                  _Float16* __restrict__ out, int n4) {
  int i = blockIdx.x * 256 + threadIdx.x;
  if (i < n4) {
    v4f x = *(const v4f*)(in + (size_t)i * 4);
    v4h y;
#pragma unroll
    for (int j = 0; j < 4; ++j) y[j] = (_Float16)x[j];
    *(v4h*)(out + (size_t)i * 4) = y;
  }
}

// Batched transpose+convert: in is (batch, R, C) f32, out is (batch, C, R) f16.
__global__ void transpose_f32_to_f16_kernel(const float* __restrict__ in,
                                            _Float16* __restrict__ out,
                                            int R, int C, int total) {
  int t = blockIdx.x * 256 + threadIdx.x;
  if (t >= total) return;
  int rc = R * C;
  int bb = t / rc;
  int rem = t - bb * rc;
  int r = rem / C;
  int c = rem - r * C;
  out[(size_t)bb * rc + (size_t)c * R + r] = (_Float16)in[t];
}

// ---------------------------------------------------------------------------
// 256x128-tile f16 WMMA GEMM, 64x64 per wave (4x4 WMMA grid), double-buffered
// LDS pipeline.   C = A(MxK) * Bt(NxK)^T (+ epilogue)
// MODE 0: Q  -> f16 out, (b,h,s,e) layout, (c+bias)*0.125
// MODE 1: K  -> f16 out, (b,h,s,e) layout, c+bias
// MODE 2: V  -> f16 out, TRANSPOSED (b,h,e,s) layout, c+bias
// MODE 3: FF1-> f16 out, row-major, relu(c+bias)
// MODE 4: FF2-> f32 out, row-major, c+bias
// ---------------------------------------------------------------------------
template <int MODE>
__global__ __launch_bounds__(256) void gemm_kernel(
    const _Float16* __restrict__ A, const _Float16* __restrict__ Bt,
    const float* __restrict__ bias, void* __restrict__ Cout,
    int M, int N, int K) {
  __shared__ _Float16 Alds[2][256 * 32];   // 16 KB per buffer
  __shared__ _Float16 Blds[2][128 * 32];   //  8 KB per buffer  (48 KB total)

  const int tid = threadIdx.x;
  const int lane = tid & 31;
  const int wv = tid >> 5;       // 8 waves
  const int l16 = lane & 15;
  const bool hi = lane >= 16;
  const int wm = wv & 3;         // 4 waves along M (64 rows each)
  const int wn = wv >> 2;        // 2 waves along N (64 cols each)
  const int mB = blockIdx.x * 256;
  const int nB = blockIdx.y * 128;

  v8f acc[4][4];
#pragma unroll
  for (int mi = 0; mi < 4; ++mi)
#pragma unroll
    for (int ni = 0; ni < 4; ++ni) acc[mi][ni] = zero8();

  // staging maps: A -> one 32-half row per thread; B -> half row per thread
  const _Float16* aRow = A + (size_t)(mB + tid) * K;
  const int bRowIdx = tid >> 1;
  const int bcol = (tid & 1) * 16;
  const _Float16* bRow = Bt + (size_t)(nB + bRowIdx) * K + bcol;
  const int aOff = tid * 32;
  const int bOff = bRowIdx * 32 + bcol;

  // prologue: stage slab 0
  {
    v8h a0 = *(const v8h*)(aRow);
    v8h a1 = *(const v8h*)(aRow + 8);
    v8h a2 = *(const v8h*)(aRow + 16);
    v8h a3 = *(const v8h*)(aRow + 24);
    v8h b0 = *(const v8h*)(bRow);
    v8h b1 = *(const v8h*)(bRow + 8);
    *(v8h*)&Alds[0][aOff]      = a0;
    *(v8h*)&Alds[0][aOff + 8]  = a1;
    *(v8h*)&Alds[0][aOff + 16] = a2;
    *(v8h*)&Alds[0][aOff + 24] = a3;
    *(v8h*)&Blds[0][bOff]      = b0;
    *(v8h*)&Blds[0][bOff + 8]  = b1;
  }
  __syncthreads();

  const int ab = hi ? 16 : 0;         // A K-chunk base (contiguous k = 16h + s)
  const int kb = hi ? 16 : 0;         // B K-chunk base (contiguous k = 16h + s)
  int cur = 0;

  for (int k0 = 0; k0 < K; k0 += 32) {
    const bool more = (k0 + 32) < K;
    // issue next slab's global loads early (overlap with WMMA below)
    v8h na0, na1, na2, na3, nb0, nb1;
    if (more) {
      na0 = *(const v8h*)(aRow + k0 + 32);
      na1 = *(const v8h*)(aRow + k0 + 40);
      na2 = *(const v8h*)(aRow + k0 + 48);
      na3 = *(const v8h*)(aRow + k0 + 56);
      nb0 = *(const v8h*)(bRow + k0 + 32);
      nb1 = *(const v8h*)(bRow + k0 + 40);
      if (k0 + 64 < K) {               // warm L2 for slab after next
        __builtin_prefetch(aRow + k0 + 64, 0, 3);
        __builtin_prefetch(bRow + k0 + 64, 0, 3);
      }
    }

    // compute on lds[cur]
    const _Float16* Al = Alds[cur];
    const _Float16* Bl = Blds[cur];
    F16Frag af[4], bf[4];
#pragma unroll
    for (int mi = 0; mi < 4; ++mi) {
      int ar = wm * 64 + mi * 16 + l16;
      af[mi].h[0] = *(const v8h*)&Al[ar * 32 + ab];
      af[mi].h[1] = *(const v8h*)&Al[ar * 32 + ab + 8];
    }
#pragma unroll
    for (int ni = 0; ni < 4; ++ni) {
      int br = wn * 64 + ni * 16 + l16;
      bf[ni].h[0] = *(const v8h*)&Bl[br * 32 + kb];
      bf[ni].h[1] = *(const v8h*)&Bl[br * 32 + kb + 8];
    }
#pragma unroll
    for (int mi = 0; mi < 4; ++mi)
#pragma unroll
      for (int ni = 0; ni < 4; ++ni)
        acc[mi][ni] = __builtin_amdgcn_wmma_f32_16x16x32_f16(
            false, af[mi].v, false, bf[ni].v, (short)0, acc[mi][ni], false, false);

    if (more) {
      *(v8h*)&Alds[cur ^ 1][aOff]      = na0;
      *(v8h*)&Alds[cur ^ 1][aOff + 8]  = na1;
      *(v8h*)&Alds[cur ^ 1][aOff + 16] = na2;
      *(v8h*)&Alds[cur ^ 1][aOff + 24] = na3;
      *(v8h*)&Blds[cur ^ 1][bOff]      = nb0;
      *(v8h*)&Blds[cur ^ 1][bOff + 8]  = nb1;
      __syncthreads();
      cur ^= 1;
    }
  }

  // epilogue
#pragma unroll
  for (int mi = 0; mi < 4; ++mi)
#pragma unroll
    for (int ni = 0; ni < 4; ++ni)
#pragma unroll
      for (int r = 0; r < 8; ++r) {
        int gm = mB + wm * 64 + mi * 16 + r + (hi ? 8 : 0);
        int gn = nB + wn * 64 + ni * 16 + l16;
        float c = acc[mi][ni][r];
        if (MODE <= 2) {
          int b = gm >> 12;            // SEQ == 4096
          int s = gm & (SEQ - 1);
          int h = gn >> 6;             // HEAD_DIM == 64
          int e = gn & 63;
          float vv = c + bias[gn];
          if (MODE == 0) vv *= 0.125f; // dh^-0.5
          _Float16* O = (_Float16*)Cout;
          if (MODE == 2)
            O[(((size_t)b * N_HEADS + h) * HEAD_DIM + e) * SEQ + s] = (_Float16)vv;
          else
            O[(((size_t)b * N_HEADS + h) * SEQ + s) * HEAD_DIM + e] = (_Float16)vv;
        } else if (MODE == 3) {
          float vv = c + bias[gn];
          vv = vv > 0.0f ? vv : 0.0f;
          ((_Float16*)Cout)[(size_t)gm * N + gn] = (_Float16)vv;
        } else {
          ((float*)Cout)[(size_t)gm * N + gn] = c + bias[gn];
        }
      }
}

// ---------------------------------------------------------------------------
// Sliding-window attention, one block per (window, batch*head).
// Branch-free: out-of-range key bases are clamped to 0; the mask stage
// forces those logits to -1e9 so their probabilities are exactly 0.
// ---------------------------------------------------------------------------
__global__ __launch_bounds__(256) void attn_kernel(
    const _Float16* __restrict__ Qh, const _Float16* __restrict__ Kh,
    const _Float16* __restrict__ Vt, float* __restrict__ Out) {
  __shared__ _Float16 attnLds[WINDOW * 2 * WINDOW];  // 128 x 256 f16 = 64 KB

  const int w = blockIdx.x;
  const int bh = blockIdx.y;
  const int b = bh >> 4;
  const int h = bh & 15;
  const int tid = threadIdx.x;
  const int lane = tid & 31;
  const int wv = tid >> 5;
  const int l16 = lane & 15;
  const bool hi = lane >= 16;

  // ----- stage 1: sim = Q * K^T over 256 keys (16 tiles), k-dim = 64 -----
  const int qrow = w * WINDOW + wv * 16 + l16;
  const _Float16* qp = Qh + ((size_t)bh * SEQ + qrow) * HEAD_DIM;
  const int ab = hi ? 16 : 0;
  F16Frag aq0, aq1;
  aq0.h[0] = *(const v8h*)(qp + ab);
  aq0.h[1] = *(const v8h*)(qp + ab + 8);
  aq1.h[0] = *(const v8h*)(qp + 32 + ab);
  aq1.h[1] = *(const v8h*)(qp + 32 + ab + 8);

  const int ko = hi ? 16 : 0;
  auto loadK = [&](int t, F16Frag& b0, F16Frag& b1) {
    int kbase = (w - 1) * WINDOW + t * 16;
    if (kbase < 0) kbase = 0;  // clamped; masked later
    const _Float16* kp = Kh + ((size_t)bh * SEQ + kbase + l16) * HEAD_DIM;
    b0.h[0] = *(const v8h*)(kp + ko);
    b0.h[1] = *(const v8h*)(kp + ko + 8);
    b1.h[0] = *(const v8h*)(kp + 32 + ko);
    b1.h[1] = *(const v8h*)(kp + 32 + ko + 8);
  };

  v8f acc[16];
  // depth-2 pipelined K-fragment fetch
  F16Frag kb0[2], kb1[2];
  loadK(0, kb0[0], kb1[0]);
  loadK(1, kb0[1], kb1[1]);
#pragma unroll
  for (int t = 0; t < 16; ++t) {
    v8f a = __builtin_amdgcn_wmma_f32_16x16x32_f16(
        false, aq0.v, false, kb0[t & 1].v, (short)0, zero8(), false, false);
    a = __builtin_amdgcn_wmma_f32_16x16x32_f16(
        false, aq1.v, false, kb1[t & 1].v, (short)0, a, false, false);
    acc[t] = a;
    if (t + 2 < 16) loadK(t + 2, kb0[t & 1], kb1[t & 1]);
  }

  // ----- mask + softmax in registers (row = 16-lane group in C layout) -----
#pragma unroll
  for (int r = 0; r < 8; ++r) {
    int qg = w * WINDOW + wv * 16 + r + (hi ? 8 : 0);
    float mx = -1e30f;
#pragma unroll
    for (int t = 0; t < 16; ++t) {
      int kg = (w - 1) * WINDOW + t * 16 + l16;  // true (unclamped) position
      float vv = (kg >= 0 && qg >= kg) ? acc[t][r] : -1e9f;
      acc[t][r] = vv;
      mx = fmaxf(mx, vv);
    }
#pragma unroll
    for (int m = 1; m < 16; m <<= 1) mx = fmaxf(mx, __shfl_xor(mx, m));
    float sum = 0.0f;
#pragma unroll
    for (int t = 0; t < 16; ++t) {
      float p = __expf(acc[t][r] - mx);
      acc[t][r] = p;
      sum += p;
    }
#pragma unroll
    for (int m = 1; m < 16; m <<= 1) sum += __shfl_xor(sum, m);
    float inv = 1.0f / sum;
    int arow = wv * 16 + r + (hi ? 8 : 0);
#pragma unroll
    for (int t = 0; t < 16; ++t)
      attnLds[arow * 256 + t * 16 + l16] = (_Float16)(acc[t][r] * inv);
  }
  __syncthreads();

  // ----- stage 2: out = P * V  (K = 256 keys, N = 64 dh) -----
  auto loadV = [&](int ks, int n, F16Frag& fb) {
    int kbase = (w - 1) * WINDOW + ks * 32;
    if (kbase < 0) kbase = 0;  // P there is exactly 0
    int e = n * 16 + l16;
    const _Float16* vp =
        Vt + ((size_t)bh * HEAD_DIM + e) * SEQ + kbase + (hi ? 16 : 0);
    fb.h[0] = *(const v8h*)(vp);
    fb.h[1] = *(const v8h*)(vp + 8);
  };

  v8f oa[4];
#pragma unroll
  for (int n = 0; n < 4; ++n) oa[n] = zero8();

  const int arow2 = wv * 16 + l16;
  F16Frag fa;                       // A fragment, reloaded per ks
  F16Frag fb[2];                    // depth-2 pipelined V fragments
  loadV(0, 0, fb[0]);
  loadV(0, 1, fb[1]);
#pragma unroll
  for (int idx = 0; idx < 32; ++idx) {
    int ks = idx >> 2;
    int n = idx & 3;
    if (n == 0) {
      int aoff = ks * 32 + (hi ? 16 : 0);
      fa.h[0] = *(const v8h*)&attnLds[arow2 * 256 + aoff];
      fa.h[1] = *(const v8h*)&attnLds[arow2 * 256 + aoff + 8];
    }
    oa[n] = __builtin_amdgcn_wmma_f32_16x16x32_f16(
        false, fa.v, false, fb[idx & 1].v, (short)0, oa[n], false, false);
    if (idx + 2 < 32) loadV((idx + 2) >> 2, (idx + 2) & 3, fb[idx & 1]);
  }

#pragma unroll
  for (int n = 0; n < 4; ++n)
#pragma unroll
    for (int r = 0; r < 8; ++r) {
      int srow = w * WINDOW + wv * 16 + r + (hi ? 8 : 0);
      int col = h * HEAD_DIM + n * 16 + l16;
      Out[((size_t)b * SEQ + srow) * D_MODEL + col] = oa[n][r];
    }
}

// ---------------------------------------------------------------------------
// Fused residual-add + LayerNorm (vectorized, shuffle-based reductions);
// optionally emits f16 copy. One block per row, 256 threads, 4 elems/thread.
// ---------------------------------------------------------------------------
__global__ __launch_bounds__(256) void ln_kernel(
    const float* __restrict__ X, const float* __restrict__ Rsd,
    const float* __restrict__ g, const float* __restrict__ bb,
    float* __restrict__ outF, _Float16* __restrict__ outH) {
  __shared__ float red[8];
  const int row = blockIdx.x;
  const int tid = threadIdx.x;
  const int lane = tid & 31;
  const int wv = tid >> 5;
  const size_t base = (size_t)row * D_MODEL + (size_t)tid * 4;

  v4f xv = *(const v4f*)(X + base);
  v4f rv = *(const v4f*)(Rsd + base);
  v4f v = xv + rv;

  float s = waveReduceAdd(v[0] + v[1] + v[2] + v[3]);
  if (lane == 0) red[wv] = s;
  __syncthreads();
  float tot = 0.0f;
#pragma unroll
  for (int i = 0; i < 8; ++i) tot += red[i];
  float mu = tot * (1.0f / D_MODEL);
  __syncthreads();

  v4f d = v - mu;
  float q = waveReduceAdd(d[0] * d[0] + d[1] * d[1] + d[2] * d[2] + d[3] * d[3]);
  if (lane == 0) red[wv] = q;
  __syncthreads();
  float qtot = 0.0f;
#pragma unroll
  for (int i = 0; i < 8; ++i) qtot += red[i];
  float rstd = rsqrtf(qtot * (1.0f / D_MODEL) + 1e-5f);

  v4f gv = *(const v4f*)(g + (size_t)tid * 4);
  v4f bv = *(const v4f*)(bb + (size_t)tid * 4);
  v4f y = d * rstd * gv + bv;
  *(v4f*)(outF + base) = y;
  if (outH) {
    v4h yh;
#pragma unroll
    for (int j = 0; j < 4; ++j) yh[j] = (_Float16)y[j];
    *(v4h*)(outH + base) = yh;
  }
}

// ---------------------------------------------------------------------------
// Launcher
// ---------------------------------------------------------------------------
extern "C" void kernel_launch(void* const* d_in, const int* in_sizes, int n_in,
                              void* d_out, int out_size, void* d_ws, size_t ws_size,
                              hipStream_t stream) {
  (void)in_sizes; (void)n_in; (void)out_size; (void)ws_size;

  const float* x     = (const float*)d_in[0];
  const float* qproj = (const float*)d_in[1];
  const float* kproj = (const float*)d_in[2];
  const float* vproj = (const float*)d_in[3];
  const float* qb    = (const float*)d_in[4];
  const float* kb    = (const float*)d_in[5];
  const float* vb    = (const float*)d_in[6];
  const float* ln1g  = (const float*)d_in[7];
  const float* ln1b  = (const float*)d_in[8];
  const float* ln2g  = (const float*)d_in[9];
  const float* ln2b  = (const float*)d_in[10];
  const float* ff1w  = (const float*)d_in[11];
  const float* ff1b  = (const float*)d_in[12];
  const float* ff2w  = (const float*)d_in[13];
  const float* ff2b  = (const float*)d_in[14];

  // workspace layout (bytes)
  constexpr size_t SZ_XH  = (size_t)BS * D_MODEL * 2;          // 32 MB
  constexpr size_t SZ_W   = (size_t)D_MODEL * D_MODEL * 2;     //  2 MB
  constexpr size_t SZ_W12 = (size_t)D_MODEL * FF_HID * 2;      //  4 MB
  constexpr size_t SZ_QKV = (size_t)BHT * SEQ * HEAD_DIM * 2;  // 32 MB
  constexpr size_t SZ_F32 = (size_t)BS * D_MODEL * 4;          // 64 MB

  char* ws = (char*)d_ws;
  size_t off = 0;
  _Float16* XH  = (_Float16*)(ws + off); off += SZ_XH;
  _Float16* WQ  = (_Float16*)(ws + off); off += SZ_W;
  _Float16* WK  = (_Float16*)(ws + off); off += SZ_W;
  _Float16* WV  = (_Float16*)(ws + off); off += SZ_W;
  _Float16* W1T = (_Float16*)(ws + off); off += SZ_W12;
  _Float16* W2T = (_Float16*)(ws + off); off += SZ_W12;
  size_t off_qh = off;
  _Float16* QH  = (_Float16*)(ws + off); off += SZ_QKV;
  _Float16* KH  = (_Float16*)(ws + off); off += SZ_QKV;
  _Float16* VT  = (_Float16*)(ws + off); off += SZ_QKV;
  size_t off_attn = off;
  float*    ATTN = (float*)(ws + off);   off += SZ_F32;
  float*    Y1F  = (float*)(ws + off);   off += SZ_F32;
  _Float16* Y1H  = (_Float16*)(ws + off); off += SZ_XH;
  // aliases: attention consumers are done before these are written
  _Float16* H1   = (_Float16*)(ws + off_qh);    // 64 MB over QH+KH
  float*    FF2O = (float*)(ws + off_attn);     // 64 MB over ATTN

  const dim3 blk(256);

  // 1) precision converts / weight transposes
  f32_to_f16_kernel<<<(BS * D_MODEL / 4 + 255) / 256, blk, 0, stream>>>(
      x, XH, BS * D_MODEL / 4);
  transpose_f32_to_f16_kernel<<<(N_HEADS * D_MODEL * HEAD_DIM + 255) / 256, blk, 0, stream>>>(
      qproj, WQ, D_MODEL, HEAD_DIM, N_HEADS * D_MODEL * HEAD_DIM);
  transpose_f32_to_f16_kernel<<<(N_HEADS * D_MODEL * HEAD_DIM + 255) / 256, blk, 0, stream>>>(
      kproj, WK, D_MODEL, HEAD_DIM, N_HEADS * D_MODEL * HEAD_DIM);
  transpose_f32_to_f16_kernel<<<(N_HEADS * D_MODEL * HEAD_DIM + 255) / 256, blk, 0, stream>>>(
      vproj, WV, D_MODEL, HEAD_DIM, N_HEADS * D_MODEL * HEAD_DIM);
  transpose_f32_to_f16_kernel<<<(D_MODEL * FF_HID + 255) / 256, blk, 0, stream>>>(
      ff1w, W1T, D_MODEL, FF_HID, D_MODEL * FF_HID);
  transpose_f32_to_f16_kernel<<<(FF_HID * D_MODEL + 255) / 256, blk, 0, stream>>>(
      ff2w, W2T, FF_HID, D_MODEL, FF_HID * D_MODEL);

  // 2) QKV projections (WMMA GEMMs)
  gemm_kernel<0><<<dim3(BS / 256, D_MODEL / 128), blk, 0, stream>>>(
      XH, WQ, qb, QH, BS, D_MODEL, D_MODEL);
  gemm_kernel<1><<<dim3(BS / 256, D_MODEL / 128), blk, 0, stream>>>(
      XH, WK, kb, KH, BS, D_MODEL, D_MODEL);
  gemm_kernel<2><<<dim3(BS / 256, D_MODEL / 128), blk, 0, stream>>>(
      XH, WV, vb, VT, BS, D_MODEL, D_MODEL);

  // 3) sliding-window attention (WMMA)
  attn_kernel<<<dim3(NW, BHT), blk, 0, stream>>>(QH, KH, VT, ATTN);

  // 4) residual + LN1 (also emits f16 activations for FFN)
  ln_kernel<<<BS, blk, 0, stream>>>(x, ATTN, ln1g, ln1b, Y1F, Y1H);

  // 5) FFN (WMMA GEMMs)
  gemm_kernel<3><<<dim3(BS / 256, FF_HID / 128), blk, 0, stream>>>(
      Y1H, W1T, ff1b, H1, BS, FF_HID, D_MODEL);
  gemm_kernel<4><<<dim3(BS / 256, D_MODEL / 128), blk, 0, stream>>>(
      H1, W2T, ff2b, FF2O, BS, D_MODEL, FF_HID);

  // 6) residual + LN2 -> final output
  ln_kernel<<<BS, blk, 0, stream>>>(Y1F, FF2O, ln2g, ln2b, (float*)d_out, nullptr);
}
